// TSPE_1915555414201
// MI455X (gfx1250) — compile-verified, matrix operand order checked by reading, not run on hardware
//
#include <hip/hip_runtime.h>
#include <hip/hip_bf16.h>

typedef __attribute__((ext_vector_type(16))) __bf16 v16bf;
typedef __attribute__((ext_vector_type(8)))  __bf16 v8bf;
typedef __attribute__((ext_vector_type(8)))  float  v8f;

#define BM 128     // block rows (8 waves x 16)
#define CH 256     // hidden width (both layers)

// ---------------------------------------------------------------------------
// Degree / norm
// ---------------------------------------------------------------------------
__global__ void gcn_deg_init(float* __restrict__ deg, int n) {
    int i = blockIdx.x * blockDim.x + threadIdx.x;
    if (i < n) deg[i] = 1.0f;                 // self-loop contributes 1
}

__global__ void gcn_deg_accum(const int* __restrict__ dst, float* __restrict__ deg, int e) {
    int i = blockIdx.x * blockDim.x + threadIdx.x;
    if (i < e) unsafeAtomicAdd(&deg[dst[i]], 1.0f);
}

__global__ void gcn_rsqrt(float* __restrict__ deg, int n) {
    int i = blockIdx.x * blockDim.x + threadIdx.x;
    if (i < n) deg[i] = rsqrtf(deg[i]);       // deg >= 1 always
}

// ---------------------------------------------------------------------------
// fp32 -> bf16 conversions (zero-fill row padding for the GEMM M edge)
// ---------------------------------------------------------------------------
__global__ void gcn_cvt_bf16(const float* __restrict__ src, __bf16* __restrict__ dst,
                             int n, int npad) {
    int i = blockIdx.x * blockDim.x + threadIdx.x;
    if (i < n)          dst[i] = (__bf16)src[i];
    else if (i < npad)  dst[i] = (__bf16)0.0f;
}

__global__ void gcn_relu_cvt_bf16(const float* __restrict__ src, __bf16* __restrict__ dst,
                                  int n, int npad) {
    int i = blockIdx.x * blockDim.x + threadIdx.x;
    if (i < n)          dst[i] = (__bf16)fmaxf(src[i], 0.0f);
    else if (i < npad)  dst[i] = (__bf16)0.0f;
}

// ---------------------------------------------------------------------------
// Pack W [K,N] f32 into fragment-major bf16:
//   Bp[((nt*ktiles + kt)*32 + lane)*16 + i],  i-th elem = W[kt*32 + 16*(i>>3) + 8*hs + (i&7)][nt*16 + l16]
// One contiguous 32B chunk per lane per (nt,kt) tile -> v16bf load in GEMM.
// ---------------------------------------------------------------------------
__global__ void gcn_pack_w(const float* __restrict__ W, __bf16* __restrict__ Bp,
                           int K, int N) {
    int t = blockIdx.x * blockDim.x + threadIdx.x;
    int ktiles = K >> 5;
    int total  = (N >> 4) * ktiles * 32;
    if (t >= total) return;
    int lane = t & 31;
    int kt   = (t >> 5) % ktiles;
    int nt   = (t >> 5) / ktiles;
    int hs   = lane >> 4;
    int n    = nt * 16 + (lane & 15);
    __bf16* o = Bp + (size_t)t * 16;
    #pragma unroll
    for (int i = 0; i < 16; ++i) {
        int k = kt * 32 + ((i >> 3) << 4) + hs * 8 + (i & 7);
        o[i] = (__bf16)W[(size_t)k * N + n];
    }
}

// ---------------------------------------------------------------------------
// LDS-free bf16 WMMA GEMM: C[M,N] = A[Mpad,K](bf16,row-major) * Bp(packed)
// 256 threads = 8 wave32; wave tile 16x64 (4 accumulators); K step 32.
// Every fragment is contiguous 128-bit loads; B panel is L2-resident.
// ---------------------------------------------------------------------------
__global__ __launch_bounds__(256) void gcn_gemm_wmma(
    const __bf16* __restrict__ A, const __bf16* __restrict__ Bp,
    float* __restrict__ C, int M, int K, int N) {

    const int lane = threadIdx.x & 31;
    const int wave = threadIdx.x >> 5;          // 0..7
    const int hs   = lane >> 4;
    const int l16  = lane & 15;
    const int arow = blockIdx.x * BM + wave * 16 + l16;   // A row this lane streams
    const int n0t  = blockIdx.y * 4;                      // first 16-col tile
    const int ktiles = K >> 5;

    union Frag { v16bf v; v8bf h[2]; };
    union Acc  { v8f  v; float e[8]; };
    Acc acc[4];
    #pragma unroll
    for (int t = 0; t < 4; ++t)
        #pragma unroll
        for (int q = 0; q < 8; ++q) acc[t].e[q] = 0.0f;

    const __bf16* aptr = A + (size_t)arow * K + hs * 8;

    for (int kt = 0; kt < ktiles; ++kt) {
        Frag a;
        a.h[0] = *(const v8bf*)(aptr);        // k = 32kt + 8hs + 0..7
        a.h[1] = *(const v8bf*)(aptr + 16);   // k = 32kt + 16 + 8hs + 0..7
        __builtin_prefetch(aptr + 32, 0, 0);  // global_prefetch_b8: next K chunk
        aptr += 32;

        #pragma unroll
        for (int nt = 0; nt < 4; ++nt) {
            Frag b;
            b.v = *(const v16bf*)(Bp + (((size_t)(n0t + nt) * ktiles + kt) * 32 + lane) * 16);
            acc[nt].v = __builtin_amdgcn_wmma_f32_16x16x32_bf16(
                false, a.v, false, b.v, (short)0, acc[nt].v, false, false);
        }
    }

    // C/D layout: m = q + 8*hs within wave strip, n = l16 within 16-col tile
    const int orow = blockIdx.x * BM + wave * 16;
    #pragma unroll
    for (int nt = 0; nt < 4; ++nt) {
        #pragma unroll
        for (int q = 0; q < 8; ++q) {
            int r = orow + q + 8 * hs;
            if (r < M) C[(size_t)r * N + (n0t + nt) * 16 + l16] = acc[nt].e[q];
        }
    }
}

// ---------------------------------------------------------------------------
// Aggregation: init with bias, edge-parallel scatter-add (native f32 atomics)
// ---------------------------------------------------------------------------
__global__ void gcn_agg_init(float* __restrict__ agg, const float* __restrict__ bias, int total) {
    int i = blockIdx.x * blockDim.x + threadIdx.x;
    if (i < total) agg[i] = bias[i & (CH - 1)];
}

__global__ void gcn_scatter(const float* __restrict__ h,
                            const int* __restrict__ src, const int* __restrict__ dst,
                            const float* __restrict__ dinv, float* __restrict__ agg,
                            int e_real, int total_threads) {
    int idx = blockIdx.x * blockDim.x + threadIdx.x;
    if (idx >= total_threads) return;
    int e = idx >> 6;               // 64 threads per edge
    int c = (idx & 63) << 2;        // 4 channels per thread
    int s, d;
    if (e < e_real) { s = src[e]; d = dst[e]; }
    else            { s = d = e - e_real; }           // synthesized self-loops
    float w = dinv[s] * dinv[d];
    float4 v = *(const float4*)(h + (size_t)s * CH + c);
    float* p = agg + (size_t)d * CH + c;
    unsafeAtomicAdd(p + 0, v.x * w);
    unsafeAtomicAdd(p + 1, v.y * w);
    unsafeAtomicAdd(p + 2, v.z * w);
    unsafeAtomicAdd(p + 3, v.w * w);
}

__global__ void gcn_relu(float* __restrict__ x, int total) {
    int i = blockIdx.x * blockDim.x + threadIdx.x;
    if (i < total) x[i] = fmaxf(x[i], 0.0f);
}

// ---------------------------------------------------------------------------
// Launch
// ---------------------------------------------------------------------------
extern "C" void kernel_launch(void* const* d_in, const int* in_sizes, int n_in,
                              void* d_out, int out_size, void* d_ws, size_t ws_size,
                              hipStream_t stream) {
    const float* x   = (const float*)d_in[0];
    const int*   ei  = (const int*)d_in[1];     // [2, E]
    const float* W1  = (const float*)d_in[2];
    const float* b1  = (const float*)d_in[3];
    const float* W2  = (const float*)d_in[4];
    const float* b2  = (const float*)d_in[5];
    float*       out = (float*)d_out;

    const int E      = in_sizes[1] / 2;         // 500000
    const int HID    = in_sizes[3];             // 256
    const int IN_CH  = in_sizes[2] / HID;       // 128
    const int NN     = in_sizes[0] / IN_CH;     // 50000
    const int MPAD   = ((NN + BM - 1) / BM) * BM;

    const int* src = ei;
    const int* dst = ei + E;

    // ---- workspace carve (32B-aligned chunks) -----------------------------
    char* w = (char*)d_ws;
    float*  h    = (float*)w;  w += (size_t)NN * CH * sizeof(float);
    float*  agg  = (float*)w;  w += (size_t)NN * CH * sizeof(float);
    float*  dinv = (float*)w;  w += (((size_t)NN * sizeof(float)) + 31) & ~(size_t)31;
    __bf16* abf  = (__bf16*)w; w += (size_t)MPAD * CH * sizeof(__bf16);  // xbf then zbf
    __bf16* Bp1  = (__bf16*)w; w += (size_t)IN_CH * HID * sizeof(__bf16);
    __bf16* Bp2  = (__bf16*)w;

    const int T = 256;
    const int totC  = NN * CH;
    const int sthreads = (E + NN) * 64;
    auto cdiv = [](int a, int b) { return (a + b - 1) / b; };

    // ---- degree / norm ----------------------------------------------------
    gcn_deg_init <<<cdiv(NN, T), T, 0, stream>>>(dinv, NN);
    gcn_deg_accum<<<cdiv(E, T),  T, 0, stream>>>(dst, dinv, E);
    gcn_rsqrt    <<<cdiv(NN, T), T, 0, stream>>>(dinv, NN);

    // ---- weight packing ---------------------------------------------------
    gcn_pack_w<<<cdiv((HID / 16) * (IN_CH / 32) * 32, T), T, 0, stream>>>(W1, Bp1, IN_CH, HID);
    gcn_pack_w<<<cdiv((HID / 16) * (HID   / 32) * 32, T), T, 0, stream>>>(W2, Bp2, HID, HID);

    dim3 gg(MPAD / BM, HID / 64);

    // ---- layer 1 ----------------------------------------------------------
    gcn_cvt_bf16 <<<cdiv(MPAD * IN_CH, T), T, 0, stream>>>(x, abf, NN * IN_CH, MPAD * IN_CH);
    gcn_gemm_wmma<<<gg, T, 0, stream>>>(abf, Bp1, h, NN, IN_CH, HID);
    gcn_agg_init <<<cdiv(totC, T), T, 0, stream>>>(agg, b1, totC);
    gcn_scatter  <<<cdiv(sthreads, T), T, 0, stream>>>(h, src, dst, dinv, agg, E, sthreads);

    // ---- layer 2 (relu fused into bf16 conversion; aggregate into d_out) --
    gcn_relu_cvt_bf16<<<cdiv(MPAD * HID, T), T, 0, stream>>>(agg, abf, totC, MPAD * HID);
    gcn_gemm_wmma<<<gg, T, 0, stream>>>(abf, Bp2, h, NN, HID, HID);
    gcn_agg_init <<<cdiv(totC, T), T, 0, stream>>>(out, b2, totC);
    gcn_scatter  <<<cdiv(sthreads, T), T, 0, stream>>>(h, src, dst, dinv, out, E, sthreads);
    gcn_relu     <<<cdiv(totC, T), T, 0, stream>>>(out, totC);
}
